// MambaBlock_12833362281153
// MI455X (gfx1250) — matrix operand order repaired and checked by reading, not compiled, not run on hardware
//
#include <hip/hip_runtime.h>
#include <hip/hip_bf16.h>

// ---------------------------------------------------------------------------
// Problem constants (from the reference)
// ---------------------------------------------------------------------------
#define BATCH   4
#define LENGTH  512
#define D_MODEL 128
#define D_STATE 256
#define D_INNER 256
#define ROWS    (BATCH * LENGTH)   // 2048 "token" rows

typedef __attribute__((ext_vector_type(16))) _Float16 v16h;
typedef __attribute__((ext_vector_type(8)))  _Float16 v8h;
typedef __attribute__((ext_vector_type(8)))  float    v8f;

// ---------------------------------------------------------------------------
// Weight packing: B[K,N] f32 row-major  ->  f16 WMMA B-fragment order.
// Fragment (tn, ks) covers N in [16tn,16tn+16), K in [32ks,32ks+32).
// Lane l = (n%16) + 16*((k%32)/16) ; element e = k%16  (contiguous k order, same as A)
// Packed addr = (((tn*(K/32)+ks)*32)+lane)*16 + e  -> each lane's 16 halves
// are 32 contiguous bytes (two b128 loads in the GEMM).
// ---------------------------------------------------------------------------
__global__ __launch_bounds__(256)
void pack_b_f16(const float* __restrict__ B, _Float16* __restrict__ Bp, int K, int N)
{
    const int i = blockIdx.x * 256 + threadIdx.x;
    if (i >= K * N) return;
    const int k = i / N, n = i % N;
    const int tn = n >> 4, ks = k >> 5;
    const int lane = (n & 15) + (((k >> 4) & 1) << 4);
    const int e = k & 15;
    const size_t dst = (((size_t)(tn * (K >> 5) + ks)) * 32 + lane) * 16 + e;
    Bp[dst] = (_Float16)B[i];
}

__global__ __launch_bounds__(256)
void f32_to_f16(const float* __restrict__ src, _Float16* __restrict__ dst, int n)
{
    const int i = blockIdx.x * 256 + threadIdx.x;
    if (i < n) dst[i] = (_Float16)src[i];
}

// ---------------------------------------------------------------------------
// WMMA GEMM: C[M,N] = Ah[M,K] @ B  (Ah f16 row-major, B pre-packed f16).
// One 16(M) x 64(N) tile per wave: 1 A fragment reused across 4 B fragments
// per K-step; K templated so the loop fully unrolls into back-to-back WMMAs.
// Optionally writes an f16 copy of C (Ch) for the next GEMM's A operand.
// ---------------------------------------------------------------------------
template <int K>
__global__ __launch_bounds__(256)
void wmma_gemm_ph(const _Float16* __restrict__ Ah, const _Float16* __restrict__ Bp,
                  float* __restrict__ C, _Float16* __restrict__ Ch,
                  int M, int N, int lda, int ldc)
{
    constexpr int KS = K / 32;
    const int lane   = threadIdx.x & 31;
    const int waveId = threadIdx.x >> 5;                  // 8 waves / block
    const int groupsN = N >> 6;                           // 64-wide N groups
    const int numWT   = (M >> 4) * groupsN;
    const int wt      = blockIdx.x * 8 + waveId;
    if (wt >= numWT) return;                              // wave-uniform

    const int M0  = (wt / groupsN) << 4;
    const int tn0 = (wt % groupsN) << 2;                  // first 16-wide N tile

    const int g = lane >> 4;                              // lane group 0/1
    const int m = lane & 15;

    const _Float16* arow = Ah + (size_t)(M0 + m) * lda;
    const _Float16* bbase = Bp + (size_t)lane * 16;       // + (tn*KS+ks)*512

    __builtin_prefetch(Bp + (size_t)tn0 * KS * 512, 0, 3);  // global_prefetch_b8

    v8f acc0 = {}, acc1 = {}, acc2 = {}, acc3 = {};

#pragma unroll
    for (int ks = 0; ks < KS; ++ks) {
        // A fragment: halves K = 16g+e (e<16) -> two b128
        v16h af;
        *((v8h*)&af)       = *(const v8h*)(arow + ks * 32 + 16 * g);
        *(((v8h*)&af) + 1) = *(const v8h*)(arow + ks * 32 + 16 * g + 8);

        const v16h bf0 = *(const v16h*)(bbase + ((size_t)(tn0 + 0) * KS + ks) * 512);
        const v16h bf1 = *(const v16h*)(bbase + ((size_t)(tn0 + 1) * KS + ks) * 512);
        const v16h bf2 = *(const v16h*)(bbase + ((size_t)(tn0 + 2) * KS + ks) * 512);
        const v16h bf3 = *(const v16h*)(bbase + ((size_t)(tn0 + 3) * KS + ks) * 512);

        acc0 = __builtin_amdgcn_wmma_f32_16x16x32_f16(false, af, false, bf0, (short)0, acc0, false, false);
        acc1 = __builtin_amdgcn_wmma_f32_16x16x32_f16(false, af, false, bf1, (short)0, acc1, false, false);
        acc2 = __builtin_amdgcn_wmma_f32_16x16x32_f16(false, af, false, bf2, (short)0, acc2, false, false);
        acc3 = __builtin_amdgcn_wmma_f32_16x16x32_f16(false, af, false, bf3, (short)0, acc3, false, false);
    }

    // D layout: vgpr r -> row M0 + r + 8g, col = 16*tile + m
#pragma unroll
    for (int r = 0; r < 8; ++r) {
        const int row = M0 + r + 8 * g;
        float* crow = C + (size_t)row * ldc + (tn0 << 4) + m;
        crow[0]  = acc0[r];
        crow[16] = acc1[r];
        crow[32] = acc2[r];
        crow[48] = acc3[r];
        if (Ch) {
            _Float16* hrow = Ch + (size_t)row * ldc + (tn0 << 4) + m;
            hrow[0]  = (_Float16)acc0[r];
            hrow[16] = (_Float16)acc1[r];
            hrow[32] = (_Float16)acc2[r];
            hrow[48] = (_Float16)acc3[r];
        }
    }
}

// ---------------------------------------------------------------------------
// Sequential selective-scan, one block per batch, state held in VGPRs.
//   thread t : d = t>>2, quarter q = t&3, owns n in [64q, 64q+64)
//   state[b,d,n] = exp(delta[b,l,d]*A[d,n]) * state + delta[b,l,d]*Bv[b,l,n]*xs[b,l,n]
//   y[b,l,d]    = sum_n state[b,d,n]*Cv[b,l,n] ;  yg_h = f16(y * silu(z))
// xp holds [xs | z] with row stride 2*D_INNER (f32, full precision scan).
// ---------------------------------------------------------------------------
__global__ __launch_bounds__(1024)
void mamba_scan(const float* __restrict__ xp,    // (B,L,512)  [xs|z]
                const float* __restrict__ delta, // (B,L,256)
                const float* __restrict__ Bv,    // (B,L,256)
                const float* __restrict__ Cv,    // (B,L,256)
                const float* __restrict__ Amat,  // (256,256)
                _Float16* __restrict__ ygh)      // (B,L,256) f16
{
    const int b  = blockIdx.x;            // 0..3
    const int t  = threadIdx.x;           // 0..1023
    const int d  = t >> 2;                // 0..255
    const int q  = t & 3;                 // n-quarter
    const int nb = q * 64;

    float a[64], s[64];
    const float* arow = Amat + (size_t)d * D_STATE + nb;
#pragma unroll
    for (int j4 = 0; j4 < 16; ++j4) {
        const float4 av = ((const float4*)arow)[j4];
        a[4*j4+0] = av.x; a[4*j4+1] = av.y; a[4*j4+2] = av.z; a[4*j4+3] = av.w;
        s[4*j4+0] = 0.f;  s[4*j4+1] = 0.f;  s[4*j4+2] = 0.f;  s[4*j4+3] = 0.f;
    }

    const size_t base  = (size_t)b * LENGTH * D_INNER;
    const size_t xbase = (size_t)b * LENGTH * (2 * D_INNER);

    for (int l = 0; l < LENGTH; ++l) {
        const size_t row  = base  + (size_t)l * D_INNER;
        const size_t xrow = xbase + (size_t)l * (2 * D_INNER);

        const float dl = delta[row + d];
        const float4* xv = (const float4*)(xp + xrow + nb);
        const float4* bv = (const float4*)(Bv + row  + nb);
        const float4* cv = (const float4*)(Cv + row  + nb);

        float acc = 0.f;
#pragma unroll
        for (int j4 = 0; j4 < 16; ++j4) {
            const float4 xw = xv[j4];
            const float4 bw = bv[j4];
            const float4 cw = cv[j4];
            const int j = 4 * j4;
            {   float dA = __expf(dl * a[j+0]);
                s[j+0] = fmaf(dA, s[j+0], dl * bw.x * xw.x);
                acc    = fmaf(s[j+0], cw.x, acc); }
            {   float dA = __expf(dl * a[j+1]);
                s[j+1] = fmaf(dA, s[j+1], dl * bw.y * xw.y);
                acc    = fmaf(s[j+1], cw.y, acc); }
            {   float dA = __expf(dl * a[j+2]);
                s[j+2] = fmaf(dA, s[j+2], dl * bw.z * xw.z);
                acc    = fmaf(s[j+2], cw.z, acc); }
            {   float dA = __expf(dl * a[j+3]);
                s[j+3] = fmaf(dA, s[j+3], dl * bw.w * xw.w);
                acc    = fmaf(s[j+3], cw.w, acc); }
        }

        acc += __shfl_xor(acc, 1, 32);
        acc += __shfl_xor(acc, 2, 32);

        if (q == 0) {
            const float z   = xp[xrow + D_INNER + d];
            const float sig = 1.f / (1.f + __expf(-z));
            ygh[row + d] = (_Float16)(acc * (z * sig));   // y * silu(z)
        }
    }
}

// ---------------------------------------------------------------------------
// Host-side orchestration (graph-capture safe: stream-only, deterministic)
// ---------------------------------------------------------------------------
extern "C" void kernel_launch(void* const* d_in, const int* in_sizes, int n_in,
                              void* d_out, int out_size, void* d_ws, size_t ws_size,
                              hipStream_t stream)
{
    (void)in_sizes; (void)n_in; (void)out_size; (void)ws_size;

    const float* x       = (const float*)d_in[0]; // (4,512,128)
    const float* W_in    = (const float*)d_in[1]; // (128,512)
    const float* W_delta = (const float*)d_in[2]; // (256,256)
    const float* W_B     = (const float*)d_in[3]; // (256,256)
    const float* W_C     = (const float*)d_in[4]; // (256,256)
    const float* W_out   = (const float*)d_in[5]; // (256,128)
    const float* Amat    = (const float*)d_in[6]; // (256,256)
    float*       out     = (float*)d_out;         // (4,512,128)

    // ---- workspace layout ----
    char* ws = (char*)d_ws;
    float*    xp     = (float*)ws;                       ws += (size_t)ROWS * 512 * 4; // f32 [xs|z]
    float*    delta  = (float*)ws;                       ws += (size_t)ROWS * 256 * 4;
    float*    Bv     = (float*)ws;                       ws += (size_t)ROWS * 256 * 4;
    float*    Cv     = (float*)ws;                       ws += (size_t)ROWS * 256 * 4;
    _Float16* x_h    = (_Float16*)ws;                    ws += (size_t)ROWS * 128 * 2;
    _Float16* xp_h   = (_Float16*)ws;                    ws += (size_t)ROWS * 512 * 2;
    _Float16* yg_h   = (_Float16*)ws;                    ws += (size_t)ROWS * 256 * 2;
    _Float16* Win_p  = (_Float16*)ws;                    ws += (size_t)128 * 512 * 2;
    _Float16* Wd_p   = (_Float16*)ws;                    ws += (size_t)256 * 256 * 2;
    _Float16* Wb_p   = (_Float16*)ws;                    ws += (size_t)256 * 256 * 2;
    _Float16* Wc_p   = (_Float16*)ws;                    ws += (size_t)256 * 256 * 2;
    _Float16* Wout_p = (_Float16*)ws;

    const dim3 blk(256);
    auto nblk = [](int n) { return dim3((n + 255) / 256); };
    auto gblk = [](int M, int N) { return dim3(((M >> 4) * (N >> 6) + 7) / 8); };

    // ---- 0) convert x to f16, pack all weights into WMMA fragment order ----
    f32_to_f16<<<nblk(ROWS * 128), blk, 0, stream>>>(x, x_h, ROWS * 128);
    pack_b_f16<<<nblk(128 * 512), blk, 0, stream>>>(W_in,    Win_p,  128, 512);
    pack_b_f16<<<nblk(256 * 256), blk, 0, stream>>>(W_delta, Wd_p,   256, 256);
    pack_b_f16<<<nblk(256 * 256), blk, 0, stream>>>(W_B,     Wb_p,   256, 256);
    pack_b_f16<<<nblk(256 * 256), blk, 0, stream>>>(W_C,     Wc_p,   256, 256);
    pack_b_f16<<<nblk(256 * 128), blk, 0, stream>>>(W_out,   Wout_p, 256, 128);

    // ---- 1) xp = x @ W_in   (also emit f16 copy for the next GEMMs) ----
    wmma_gemm_ph<128><<<gblk(ROWS, 512), blk, 0, stream>>>(
        x_h, Win_p, xp, xp_h, ROWS, 512, 128, 512);

    // ---- 2) delta/Bv/Cv = xs @ {W_delta, W_B, W_C}  (xs = xp cols [0,256)) ----
    wmma_gemm_ph<256><<<gblk(ROWS, 256), blk, 0, stream>>>(
        xp_h, Wd_p, delta, (_Float16*)nullptr, ROWS, 256, 512, 256);
    wmma_gemm_ph<256><<<gblk(ROWS, 256), blk, 0, stream>>>(
        xp_h, Wb_p, Bv, (_Float16*)nullptr, ROWS, 256, 512, 256);
    wmma_gemm_ph<256><<<gblk(ROWS, 256), blk, 0, stream>>>(
        xp_h, Wc_p, Cv, (_Float16*)nullptr, ROWS, 256, 512, 256);

    // ---- 3) sequential scan + SiLU gating (state in VGPRs), f16 out ----
    mamba_scan<<<dim3(BATCH), dim3(1024), 0, stream>>>(xp, delta, Bv, Cv, Amat, yg_h);

    // ---- 4) out = yg @ W_out ----
    wmma_gemm_ph<256><<<gblk(ROWS, 128), blk, 0, stream>>>(
        yg_h, Wout_p, out, (_Float16*)nullptr, ROWS, 128, 256, 128);
}
